// GE2ELoss_5978594476766
// MI455X (gfx1250) — matrix operand order repaired and checked by reading, not compiled, not run on hardware
//
#include <hip/hip_runtime.h>
#include <hip/hip_bf16.h>
#include <stdint.h>

#define N_SPK 1024
#define M_UTT 32
#define E_DIM 512
#define ROWS  (N_SPK * M_UTT)   // 32768
#define TILE_ELEMS (16 * E_DIM) // one B tile: 16 cols x 512 K = 8192 bf16 = 16KB

typedef __attribute__((ext_vector_type(16))) __bf16 v16bf;
typedef __attribute__((ext_vector_type(8)))  __bf16 v8bf;
typedef __attribute__((ext_vector_type(8)))  float  v8f;
typedef __attribute__((ext_vector_type(4)))  unsigned int v4u;
typedef __attribute__((ext_vector_type(8)))  int v8i;
typedef __attribute__((ext_vector_type(4)))  int v4i;

#if defined(__AMDGCN__) && __has_builtin(__builtin_amdgcn_tensor_load_to_lds) && \
    __has_builtin(__builtin_amdgcn_s_wait_tensorcnt)
#define USE_TDM 1
#else
#define USE_TDM 0
#endif

__device__ __forceinline__ unsigned short f2bf(float f) {
  unsigned int u = __float_as_uint(f);
  u += 0x7FFFu + ((u >> 16) & 1u);       // round-to-nearest-even
  return (unsigned short)(u >> 16);
}

#if USE_TDM
// Issue one TDM descriptor: contiguous 8192 x bf16 tile (16KB) -> LDS.
// D# layout per CDNA5 ISA 8.3/8.4: group0 = {ctrl, lds_addr, global_addr, type},
// group1 = {data_size=2B, tensor_dim0=8192, tensor_dim1=1, tile_dim0=8192,
//           tile_dim1=1, dim0_stride=8192}; groups 2/3 unused (zero).
__device__ __forceinline__ void tdm_load_tile(const __bf16* gsrc, unsigned lds_addr) {
  unsigned long long ga = (unsigned long long)(uintptr_t)gsrc;
  v4u g0;
  g0.x = 0x1u;                                          // count=1 (valid), user mode
  g0.y = lds_addr;                                      // LDS byte address
  g0.z = (unsigned)(ga & 0xFFFFFFFFu);                  // global_addr[31:0]
  g0.w = (unsigned)((ga >> 32) & 0x1FFFFFFu) | (2u << 30); // addr[56:32] | type=2
  v8i g1;
  g1[0] = 0x00010000;                 // data_size=1 (2 bytes), no multicast/flags
  g1[1] = (int)((TILE_ELEMS & 0xFFFF) << 16);   // tensor_dim0[15:0] @ bits 63:48
  g1[2] = (int)(1u << 16);                      // tensor_dim1=1 @ bits 111:80
  g1[3] = (int)((TILE_ELEMS & 0xFFFF) << 16);   // tile_dim0 @ bits 127:112
  g1[4] = 1;                                    // tile_dim1=1, tile_dim2=0
  g1[5] = TILE_ELEMS;                           // tensor_dim0_stride[31:0]
  g1[6] = 0;
  g1[7] = 0;
  v4i gz = {0, 0, 0, 0};
#if __clang_major__ >= 23
  v8i gz8 = {0, 0, 0, 0, 0, 0, 0, 0};
  __builtin_amdgcn_tensor_load_to_lds(g0, g1, gz, gz, gz8, 0);
#else
  __builtin_amdgcn_tensor_load_to_lds(g0, g1, gz, gz, 0);
#endif
}
#endif

__global__ void init_out(float* out) { out[0] = 0.0f; }

// One block per speaker n. Computes:
//   Bbf[n]      = unit(sum_m e[n,m]) in bf16         (GEMM B matrix = c_u)
//   Abf[n*32+m] = unit(e[n,m]) in bf16               (GEMM A matrix = e_u)
//   diag[n*32+m]= dot(e_u, unit(s - e))  (fp32)
//   out        -= sum of pos = w*(diag+1e-6)+b
__global__ void prep_kernel(const float* __restrict__ emb,
                            const float* __restrict__ wp,
                            const float* __restrict__ bp,
                            unsigned short* __restrict__ Abf,
                            unsigned short* __restrict__ Bbf,
                            float* __restrict__ diag,
                            float* __restrict__ out) {
  __shared__ float s_lds[E_DIM];
  __shared__ float red8[8];
  __shared__ float sinv_sh;
  const int n    = blockIdx.x;
  const int tid  = threadIdx.x;        // 256 threads = 8 waves (wave32)
  const int wave = tid >> 5;
  const int lane = tid & 31;
  const float* base = emb + (size_t)n * M_UTT * E_DIM;

  // phase 1: s = sum over m (each thread owns 2 components)
  const int c0 = tid * 2;
  float s0 = 0.f, s1 = 0.f;
  for (int m = 0; m < M_UTT; ++m) {
    s0 += base[m * E_DIM + c0];
    s1 += base[m * E_DIM + c0 + 1];
  }
  s_lds[c0]     = s0;
  s_lds[c0 + 1] = s1;

  // phase 2: ||s|| -> c_u = unit(s) (unit(mean) == unit(sum))
  float p = s0 * s0 + s1 * s1;
  p += __shfl_xor(p, 1, 32);
  p += __shfl_xor(p, 2, 32);
  p += __shfl_xor(p, 4, 32);
  p += __shfl_xor(p, 8, 32);
  p += __shfl_xor(p, 16, 32);
  if (lane == 0) red8[wave] = p;
  __syncthreads();
  if (tid == 0) {
    float tot = 0.f;
    for (int k = 0; k < 8; ++k) tot += red8[k];
    sinv_sh = 1.0f / fmaxf(sqrtf(tot), 1e-8f);
  }
  __syncthreads();
  const float ci = sinv_sh;
  Bbf[(size_t)n * E_DIM + c0]     = f2bf(s0 * ci);
  Bbf[(size_t)n * E_DIM + c0 + 1] = f2bf(s1 * ci);

  const float w = wp[0], bsc = bp[0];

  // phase 3: each wave handles 4 utterances, lane owns 16 contiguous comps
  for (int j = 0; j < 4; ++j) {
    const int m = wave * 4 + j;
    const float* v  = base + m * E_DIM + lane * 16;
    const float* sp = s_lds + lane * 16;
    float vloc[16];
    float vv = 0.f, vt = 0.f, tt = 0.f;
    #pragma unroll
    for (int i = 0; i < 16; ++i) {
      float x = v[i];
      float t = sp[i] - x;     // excl centroid direction (scale-free)
      vv += x * x; vt += x * t; tt += t * t;
      vloc[i] = x;
    }
    vv += __shfl_xor(vv, 1, 32); vt += __shfl_xor(vt, 1, 32); tt += __shfl_xor(tt, 1, 32);
    vv += __shfl_xor(vv, 2, 32); vt += __shfl_xor(vt, 2, 32); tt += __shfl_xor(tt, 2, 32);
    vv += __shfl_xor(vv, 4, 32); vt += __shfl_xor(vt, 4, 32); tt += __shfl_xor(tt, 4, 32);
    vv += __shfl_xor(vv, 8, 32); vt += __shfl_xor(vt, 8, 32); tt += __shfl_xor(tt, 8, 32);
    vv += __shfl_xor(vv, 16, 32); vt += __shfl_xor(vt, 16, 32); tt += __shfl_xor(tt, 16, 32);
    const float inv_v = 1.0f / fmaxf(sqrtf(vv), 1e-8f);
    const float inv_t = 1.0f / fmaxf(sqrtf(tt), 1e-8f);
    const size_t row = (size_t)n * M_UTT + m;
    unsigned short* arow = Abf + row * E_DIM + lane * 16;
    #pragma unroll
    for (int i = 0; i < 16; ++i) arow[i] = f2bf(vloc[i] * inv_v);
    if (lane == 0) {
      float d = vt * inv_v * inv_t;
      diag[row] = d;
      float pos = w * (d + 1e-6f) + bsc;
      atomicAdd(out, -pos);
    }
  }
}

// 256 threads = 8 waves, one block per 128 rows.
// A panels (16 rows x 512 K per wave) are register-resident (v16bf afrag[16]).
// B tiles (16KB) are double-buffered through LDS via the Tensor Data Mover
// (one DMA per tile, issued by wave 0, overlapped with compute on the other
// buffer). Epilogue fuses diag substitution + exp/LSE into the loss scalar.
__global__ __launch_bounds__(256, 1)
void gemm_lse_kernel(const __bf16* __restrict__ A,
                     const __bf16* __restrict__ B,
                     const float* __restrict__ diag,
                     const float* __restrict__ wp,
                     const float* __restrict__ bp,
                     float* __restrict__ out) {
  __shared__ __bf16 Blds[2][TILE_ELEMS];          // 2 x 16KB
  const int tid      = threadIdx.x;
  const int wave     = tid >> 5;
  const int lane     = tid & 31;
  const int laneRow  = lane & 15;    // A row in tile / B col in tile (N index)
  const int laneHalf = lane >> 4;    // K-group select per ISA 16-bit layouts
  const int rowBlock = blockIdx.x * 128;
  const int rowTile  = rowBlock + wave * 16;
  const int nspk     = rowTile >> 5; // single speaker per 16-row tile
  const float w = wp[0], bsc = bp[0];

  // Load this wave's A panel into registers: 16 k-steps x 16 bf16 per lane.
  // A 16x32 bf16 fragment layout: lane holds K contiguous {base..base+15},
  // base = laneHalf*16 (matches B fragment).
  v16bf afrag[16];
  {
    const __bf16* Arow = A + (size_t)(rowTile + laneRow) * E_DIM + laneHalf * 16;
    #pragma unroll
    for (int ks = 0; ks < 16; ++ks) {
      v8bf lo = *(const v8bf*)(Arow + ks * 32);
      v8bf hi = *(const v8bf*)(Arow + ks * 32 + 8);
      #pragma unroll
      for (int i = 0; i < 8; ++i) { afrag[ks][i] = lo[i]; afrag[ks][8 + i] = hi[i]; }
    }
  }

  float rs[8];
  #pragma unroll
  for (int i = 0; i < 8; ++i) rs[i] = 0.f;

#if USE_TDM
  if (wave == 0) tdm_load_tile(B, (unsigned)(uintptr_t)&Blds[0][0]);
#else
  {
    const v8bf* src = (const v8bf*)B;
    v8bf* dst = (v8bf*)(&Blds[0][0]);
    #pragma unroll
    for (int it = 0; it < 4; ++it)
      dst[it * 256 + tid] = src[it * 256 + tid];
  }
  __syncthreads();
#endif

  for (int ct = 0; ct < 64; ++ct) {
    const int buf = ct & 1;
#if USE_TDM
    if (wave == 0) __builtin_amdgcn_s_wait_tensorcnt(0);  // buf ct ready
    __syncthreads();
    // issue DMA for tile ct+1 now; it overlaps this tile's 16 WMMAs
    if (wave == 0 && ct + 1 < 64)
      tdm_load_tile(B + (size_t)(ct + 1) * TILE_ELEMS,
                    (unsigned)(uintptr_t)&Blds[buf ^ 1][0]);
#else
    if (ct + 1 < 64) {
      const v8bf* src = (const v8bf*)(B + (size_t)(ct + 1) * TILE_ELEMS);
      v8bf* dst = (v8bf*)(&Blds[buf ^ 1][0]);
      #pragma unroll
      for (int it = 0; it < 4; ++it)
        dst[it * 256 + tid] = src[it * 256 + tid];
    }
#endif

    // B 32x16 bf16 fragment: lane = col, K contiguous [laneHalf*16 .. +15].
    // Pipeline the LDS fragment load one k-step ahead of the WMMA.
    const __bf16* Bbase = &Blds[buf][laneRow * E_DIM + laneHalf * 16];
    v8f acc = {};
    v16bf bv = *(const v16bf*)(Bbase);
    #pragma unroll
    for (int ks = 0; ks < 16; ++ks) {
      v16bf bnext = bv;
      if (ks < 15) bnext = *(const v16bf*)(Bbase + (ks + 1) * 32);
      acc = __builtin_amdgcn_wmma_f32_16x16x32_bf16(false, afrag[ks], false, bv,
                                                    (short)0, acc, false, false);
      bv = bnext;
    }

    // C layout: VGPR i -> row (i + laneHalf*8), col = laneRow
    const int col = ct * 16 + laneRow;
    const bool isDiag = (col == nspk);
    #pragma unroll
    for (int i = 0; i < 8; ++i) {
      float c = acc[i];
      if (isDiag) c = diag[rowTile + laneHalf * 8 + i];
      rs[i] += __expf(w * (c + 1e-6f) + bsc);
    }
    __syncthreads();   // all reads of buf done before it is refilled at ct+2
  }

  // reduce over the 16 lanes of each half -> full row sums
  #pragma unroll
  for (int i = 0; i < 8; ++i) {
    float v = rs[i];
    v += __shfl_xor(v, 1, 32);
    v += __shfl_xor(v, 2, 32);
    v += __shfl_xor(v, 4, 32);
    v += __shfl_xor(v, 8, 32);
    rs[i] = v;
  }
  if (laneRow == 0) {                 // lanes 0 and 16
    float lsum = 0.f;
    #pragma unroll
    for (int i = 0; i < 8; ++i)
      lsum += __logf(rs[i] + 1e-6f);
    atomicAdd(out, lsum);
  }
}

extern "C" void kernel_launch(void* const* d_in, const int* in_sizes, int n_in,
                              void* d_out, int out_size, void* d_ws, size_t ws_size,
                              hipStream_t stream) {
  const float* emb = (const float*)d_in[0];
  const float* wp  = (const float*)d_in[1];
  const float* bp  = (const float*)d_in[2];
  float* out = (float*)d_out;

  // workspace layout: A bf16 (32MB) | B bf16 (1MB) | diag fp32 (128KB)
  unsigned short* Abf = (unsigned short*)d_ws;
  unsigned short* Bbf = Abf + (size_t)ROWS * E_DIM;
  float* diag = (float*)(Bbf + (size_t)N_SPK * E_DIM);

  init_out<<<1, 1, 0, stream>>>(out);
  prep_kernel<<<N_SPK, 256, 0, stream>>>(emb, wp, bp, Abf, Bbf, diag, out);
  gemm_lse_kernel<<<ROWS / 128, 256, 0, stream>>>(
      (const __bf16*)Abf, (const __bf16*)Bbf, diag, wp, bp, out);
}
